// LlamaModel_38671885533927
// MI455X (gfx1250) — compile-verified, matrix operand order checked by reading, not run on hardware
//
#include <hip/hip_runtime.h>
#include <hip/hip_bf16.h>

// ---------------------------------------------------------------------------
// Llama-2-ish forward, MI455X (gfx1250): bf16 WMMA GEMMs (double-buffered LDS)
// + flash attention with TDM (tensor_load_to_lds) K/V tile staging.
// ---------------------------------------------------------------------------

typedef __bf16 bf16;
typedef __attribute__((ext_vector_type(16))) __bf16 v16bf;
typedef __attribute__((ext_vector_type(8)))  __bf16 v8bf;
typedef __attribute__((ext_vector_type(4)))  __bf16 v4bf;
typedef __attribute__((ext_vector_type(8)))  float  v8f;
typedef __attribute__((ext_vector_type(4)))  unsigned int u32x4;
typedef __attribute__((ext_vector_type(8)))  int i32x8;
typedef __attribute__((ext_vector_type(4)))  int i32x4;

#define S_   2048
#define H_   2048
#define NH_  16
#define NKV_ 8
#define HD_  128
#define KVW  (NKV_ * HD_)   // 1024
#define I_   8192
#define L_   2
#define EPS_ 1e-5f
#define QSCALE 0.08838834764831845f   // HD^-0.5
#define LOG2_THETA_OVER_64 0.2076205059433476f  // log2(10000)/64

// ---------------- WMMA helpers ----------------------------------------------

static __device__ __forceinline__ v8f zero8() {
  v8f z;
#pragma unroll
  for (int i = 0; i < 8; ++i) z[i] = 0.0f;
  return z;
}

static __device__ __forceinline__ v16bf mk16(v8bf lo, v8bf hi) {
  return __builtin_shufflevector(lo, hi, 0, 1, 2, 3, 4, 5, 6, 7,
                                 8, 9, 10, 11, 12, 13, 14, 15);
}

// Fragment load: lane holds two contiguous 8-half runs at +0 and +16 elements
// (16-bit A/B matrix layout, ISA 7.12.2). 16-byte aligned by construction.
static __device__ __forceinline__ v16bf ldfrag(const bf16* p) {
  v8bf lo = *(const v8bf*)(p);
  v8bf hi = *(const v8bf*)(p + 16);
  return mk16(lo, hi);
}

static __device__ __forceinline__ v8f wmma_bf16(v16bf a, v16bf b, v8f c) {
  // (neg_a, A, neg_b, B, c_mod, C, reuse_a, reuse_b)
  return __builtin_amdgcn_wmma_f32_16x16x32_bf16(false, a, false, b,
                                                 (short)0, c, false, false);
}

// ---------------- TDM helper -------------------------------------------------
// 2D bf16 tile DMA global->LDS via Tensor Data Mover (CDNA5 ISA-8).
// D# group0: count=1 | lds_addr | global_addr | type=2.
// D# group1: data_size=1 (2B), tensor_dim0/1 == tile_dim0/1 (no OOB),
//            tensor_dim0_stride = row pitch in elements, 2D => tile_dim2=0.
// This toolchain exposes the 6-arg builtin:
//   (u32x4 g0, i32x8 g1, i32x4 g2, i32x4 g3, i32x8 extra, i32 cpol)

static __device__ __forceinline__ unsigned lds_off(const void* p) {
  // generic pointers into the LDS aperture carry the wave-relative LDS byte
  // address in the low 32 bits (ISA 10.2 aperture mapping)
  return (unsigned)(unsigned long long)p;
}

static __device__ __forceinline__ void tdm_load_2d_bf16(
    unsigned lds_addr, const void* gptr,
    unsigned dim0, unsigned dim1, unsigned stride0_elems) {
  unsigned long long ga = (unsigned long long)gptr;
  u32x4 g0;
  g0[0] = 1u;                                          // count=1, user mode
  g0[1] = lds_addr;                                    // lds_addr (bytes)
  g0[2] = (unsigned)(ga & 0xFFFFFFFFu);                // global_addr[31:0]
  g0[3] = (unsigned)((ga >> 32) & 0x1FFFFFFu) | (2u << 30);  // addr hi | type=2
  i32x8 g1;
  g1[0] = (int)(1u << 16);                             // workgroup_mask=0, data_size=1
  g1[1] = (int)((dim0 & 0xFFFFu) << 16);               // tensor_dim0[15:0] @63:48
  g1[2] = (int)((dim0 >> 16) | ((dim1 & 0xFFFFu) << 16));  // dim0 hi | dim1 lo
  g1[3] = (int)((dim1 >> 16) | ((dim0 & 0xFFFFu) << 16));  // dim1 hi | tile_dim0
  g1[4] = (int)(dim1 & 0xFFFFu);                       // tile_dim1 | tile_dim2=0
  g1[5] = (int)stride0_elems;                          // tensor_dim0_stride lo
  g1[6] = 0;                                           // stride0 hi | stride1 lo
  g1[7] = 0;                                           // stride1 hi
  i32x4 z4;
  z4[0] = 0; z4[1] = 0; z4[2] = 0; z4[3] = 0;
  i32x8 z8;
#pragma unroll
  for (int i = 0; i < 8; ++i) z8[i] = 0;
  __builtin_amdgcn_tensor_load_to_lds(g0, g1, z4, z4, z8, 0);
}

// ---------------- GEMM: C[M,N] = A[M,K] x B[K,N] (+ optional residual) ------

#define BM 128
#define BN 128
#define BK 32
#define LDT 40  // LDS row stride in halves: 80B = 16B-aligned, bank-skewed

__global__ __launch_bounds__(256) void k_gemm(const float* __restrict__ A,
                                              const float* __restrict__ Bm,
                                              const float* __restrict__ addsrc,
                                              float* __restrict__ C,
                                              int M, int N, int K) {
  __shared__ bf16 As[2][BM * LDT];
  __shared__ bf16 Bs[2][BN * LDT];
  const int t    = threadIdx.x;
  const int lane = t & 31;
  const int w    = t >> 5;
  const int lr   = lane & 15;
  const int gh   = lane >> 4;
  const size_t bm = (size_t)blockIdx.y * BM;
  const size_t bn = (size_t)blockIdx.x * BN;
  const int wm = (w >> 1) * 32;  // 4 waves along M
  const int wn = (w & 1) * 64;   // 2 waves along N

  v8f acc[2][4];
#pragma unroll
  for (int mt = 0; mt < 2; ++mt)
#pragma unroll
    for (int nt = 0; nt < 4; ++nt) acc[mt][nt] = zero8();

  // Phase-split, vectorized staging: all global_load_b128 issue first (so they
  // pipeline), then convert+store to LDS.
  auto load_tile = [&](int k0, int buf) {
    float4 av[4], bv[4];
#pragma unroll
    for (int i = 0; i < 4; ++i) {              // A: 128x32 f32, 1024 float4s
      int idx = t + (i << 8);
      int r = idx >> 3, c = idx & 7;
      av[i] = *(const float4*)&A[(bm + r) * K + k0 + c * 4];
    }
#pragma unroll
    for (int i = 0; i < 4; ++i) {              // B: 32x128 f32, 1024 float4s
      int idx = t + (i << 8);
      int n4 = idx & 31, kk = idx >> 5;
      bv[i] = *(const float4*)&Bm[(size_t)(k0 + kk) * N + bn + n4 * 4];
    }
#pragma unroll
    for (int i = 0; i < 4; ++i) {              // A -> LDS row-major, b64 stores
      int idx = t + (i << 8);
      int r = idx >> 3, c = idx & 7;
      v4bf o;
      o[0] = (bf16)av[i].x; o[1] = (bf16)av[i].y;
      o[2] = (bf16)av[i].z; o[3] = (bf16)av[i].w;
      *(v4bf*)&As[buf][r * LDT + c * 4] = o;   // byte off = r*80 + c*8, 8B align
    }
#pragma unroll
    for (int i = 0; i < 4; ++i) {              // B -> LDS transposed [n][k]
      int idx = t + (i << 8);
      int n4 = idx & 31, kk = idx >> 5;
      Bs[buf][(n4 * 4 + 0) * LDT + kk] = (bf16)bv[i].x;
      Bs[buf][(n4 * 4 + 1) * LDT + kk] = (bf16)bv[i].y;
      Bs[buf][(n4 * 4 + 2) * LDT + kk] = (bf16)bv[i].z;
      Bs[buf][(n4 * 4 + 3) * LDT + kk] = (bf16)bv[i].w;
    }
  };

  load_tile(0, 0);
  __syncthreads();

  const int nk = K / BK;
  for (int kt = 0; kt < nk; ++kt) {
    const int cur = kt & 1;
    if (kt + 2 < nk)  // global_prefetch_b8 two tiles ahead
      __builtin_prefetch(&Bm[(size_t)((kt + 2) * BK) * N + bn + (t & 127)], 0, 1);
    if (kt + 1 < nk) load_tile((kt + 1) * BK, cur ^ 1);

    v16bf af[2], bfr[4];
#pragma unroll
    for (int mt = 0; mt < 2; ++mt)
      af[mt] = ldfrag(&As[cur][(wm + mt * 16 + lr) * LDT + gh * 8]);
#pragma unroll
    for (int nt = 0; nt < 4; ++nt)
      bfr[nt] = ldfrag(&Bs[cur][(wn + nt * 16 + lr) * LDT + gh * 8]);
#pragma unroll
    for (int mt = 0; mt < 2; ++mt)
#pragma unroll
      for (int nt = 0; nt < 4; ++nt)
        acc[mt][nt] = wmma_bf16(af[mt], bfr[nt], acc[mt][nt]);
    __syncthreads();
  }

  // Epilogue. C-frag: VGPR j -> row = j + 8*(lane>>4), col = lane&15.
  // Uniform branch on residual so loads batch instead of per-element branches.
  if (addsrc) {
#pragma unroll
    for (int mt = 0; mt < 2; ++mt)
#pragma unroll
      for (int nt = 0; nt < 4; ++nt)
#pragma unroll
        for (int j = 0; j < 8; ++j) {
          size_t row = bm + wm + mt * 16 + gh * 8 + j;
          size_t col = bn + wn + nt * 16 + lr;
          C[row * N + col] = acc[mt][nt][j] + addsrc[row * N + col];
        }
  } else {
#pragma unroll
    for (int mt = 0; mt < 2; ++mt)
#pragma unroll
      for (int nt = 0; nt < 4; ++nt)
#pragma unroll
        for (int j = 0; j < 8; ++j) {
          size_t row = bm + wm + mt * 16 + gh * 8 + j;
          size_t col = bn + wn + nt * 16 + lr;
          C[row * N + col] = acc[mt][nt][j];
        }
  }
}

// ---------------- Elementwise / norm kernels --------------------------------

__global__ __launch_bounds__(256) void k_embed(const int* __restrict__ ids,
                                               const float* __restrict__ emb,
                                               float* __restrict__ x) {
  int s = blockIdx.x;
  int id = ids[s];
  for (int i = threadIdx.x; i < H_; i += 256)
    x[(size_t)s * H_ + i] = emb[(size_t)id * H_ + i];
}

__global__ __launch_bounds__(256) void k_rmsnorm(const float* __restrict__ x,
                                                 const float* __restrict__ w,
                                                 float* __restrict__ y) {
  __shared__ float red[256];
  int row = blockIdx.x;
  const float* xr = x + (size_t)row * H_;
  float ss = 0.0f;
  for (int i = threadIdx.x; i < H_; i += 256) {
    float v = xr[i];
    ss += v * v;
  }
  red[threadIdx.x] = ss;
  __syncthreads();
  for (int s = 128; s > 0; s >>= 1) {
    if (threadIdx.x < s) red[threadIdx.x] += red[threadIdx.x + s];
    __syncthreads();
  }
  float rs = rsqrtf(red[0] * (1.0f / H_) + EPS_);
  for (int i = threadIdx.x; i < H_; i += 256)
    y[(size_t)row * H_ + i] = w[i] * xr[i] * rs;
}

// RoPE + f32->bf16 for Q (folds softmax scale into Q)
__global__ __launch_bounds__(256) void k_rope_q(const float* __restrict__ q,
                                                bf16* __restrict__ qb) {
  int idx = blockIdx.x * 256 + threadIdx.x;  // S*NH*64 threads
  int d = idx & 63;
  int h = (idx >> 6) & (NH_ - 1);
  int s = idx >> 10;
  const float* base = q + (size_t)s * H_ + h * HD_;
  float x1 = base[d], x2 = base[d + 64];
  float fr = (float)s * exp2f(-(float)d * LOG2_THETA_OVER_64);
  float sn, cs;
  __sincosf(fr, &sn, &cs);
  bf16* ob = qb + (size_t)s * H_ + h * HD_;
  ob[d]      = (bf16)((x1 * cs - x2 * sn) * QSCALE);
  ob[d + 64] = (bf16)((x2 * cs + x1 * sn) * QSCALE);
}

// RoPE + f32->bf16 for K
__global__ __launch_bounds__(256) void k_rope_k(const float* __restrict__ k,
                                                bf16* __restrict__ kb) {
  int idx = blockIdx.x * 256 + threadIdx.x;  // S*NKV*64 threads
  int d = idx & 63;
  int h = (idx >> 6) & (NKV_ - 1);
  int s = idx >> 9;
  const float* base = k + (size_t)s * KVW + h * HD_;
  float x1 = base[d], x2 = base[d + 64];
  float fr = (float)s * exp2f(-(float)d * LOG2_THETA_OVER_64);
  float sn, cs;
  __sincosf(fr, &sn, &cs);
  bf16* ob = kb + (size_t)s * KVW + h * HD_;
  ob[d]      = (bf16)(x1 * cs - x2 * sn);
  ob[d + 64] = (bf16)(x2 * cs + x1 * sn);
}

// V: f32 [s][kv*HD+d] -> bf16 transposed [kv*HD+d][s] so P@V tiles are
// contiguous along the key (contraction) axis.
__global__ __launch_bounds__(256) void k_vT(const float* __restrict__ v,
                                            bf16* __restrict__ vbT) {
  int idx = blockIdx.x * 256 + threadIdx.x;  // S*KVW threads
  int c = idx & (KVW - 1);
  int s = idx >> 10;
  vbT[(size_t)c * S_ + s] = (bf16)v[(size_t)s * KVW + c];
}

__global__ __launch_bounds__(256) void k_silumul(float* __restrict__ g,
                                                 const float* __restrict__ u) {
  size_t i = (size_t)blockIdx.x * 256 + threadIdx.x;  // S*I threads
  float gv = g[i];
  g[i] = gv * (1.0f / (1.0f + __expf(-gv))) * u[i];
}

// ---------------- Flash attention (one wave per head x 16 query rows) -------
// K/V tiles are staged into LDS with the Tensor Data Mover, double-buffered so
// tile kt+1's DMA overlaps tile kt's WMMAs; sync via TENSORcnt.

__global__ __launch_bounds__(32) void k_attn(const bf16* __restrict__ qb,
                                             const bf16* __restrict__ kb,
                                             const bf16* __restrict__ vbT,
                                             float* __restrict__ o) {
  __shared__ bf16 Kt[2][32 * 128];  // [key][d], rows contiguous (8KB each)
  __shared__ bf16 Vt[2][128 * 32];  // [d][key], rows contiguous (8KB each)
  __shared__ bf16 P[16 * 32];       // prob tile reformat (C-layout -> A-layout)
  const int lane = threadIdx.x;
  const int lr = lane & 15;
  const int gh = lane >> 4;
  const int head = blockIdx.x;
  const int qrow = blockIdx.y * 16;
  const int kv = head >> 1;  // GQA: rep = NH/NKV = 2

  const bf16* kbase = kb + (size_t)kv * HD_;
  const bf16* vbase = vbT + (size_t)kv * HD_ * S_;

  // Q fragments: A 16x32 per 32-wide slice of HD=128 (direct global loads)
  v16bf qf[4];
  {
    const bf16* qp = qb + (size_t)(qrow + lr) * H_ + head * HD_;
#pragma unroll
    for (int d = 0; d < 4; ++d) qf[d] = ldfrag(qp + d * 32 + gh * 8);
  }

  v8f oacc[8];
#pragma unroll
  for (int c = 0; c < 8; ++c) oacc[c] = zero8();
  float mi[8], li[8];
#pragma unroll
  for (int j = 0; j < 8; ++j) { mi[j] = -1e30f; li[j] = 0.0f; }

  // TDM: K tile = 32 rows x 128 contiguous halves (pitch KVW); V tile = 128
  // rows x 32 contiguous halves (pitch S).
  tdm_load_2d_bf16(lds_off(&Kt[0][0]), kbase, 128, 32, KVW);
  tdm_load_2d_bf16(lds_off(&Vt[0][0]), vbase, 32, 128, S_);

  const int NT = S_ / 32;
  for (int kt = 0; kt < NT; ++kt) {
    const int cur = kt & 1;
    __builtin_amdgcn_s_wait_tensorcnt(0);  // tiles for kt are in LDS
    if (kt + 1 < NT) {
      tdm_load_2d_bf16(lds_off(&Kt[cur ^ 1][0]),
                       kbase + (size_t)((kt + 1) * 32) * KVW, 128, 32, KVW);
      tdm_load_2d_bf16(lds_off(&Vt[cur ^ 1][0]),
                       vbase + (kt + 1) * 32, 32, 128, S_);
    }

    // ---- scores: S = Q K^T, 16x32 as two 16x16 C-frags -----------------
    v8f s0 = zero8(), s1 = zero8();
#pragma unroll
    for (int d = 0; d < 4; ++d) {
      v16bf b0 = ldfrag(&Kt[cur][lr * 128 + d * 32 + gh * 8]);
      v16bf b1 = ldfrag(&Kt[cur][(16 + lr) * 128 + d * 32 + gh * 8]);
      s0 = wmma_bf16(qf[d], b0, s0);
      s1 = wmma_bf16(qf[d], b1, s1);
    }
    // ---- online softmax (rows live across lanes sharing lane>>4) -------
#pragma unroll
    for (int j = 0; j < 8; ++j) {
      float a = s0[j], b = s1[j];
      float mx = fmaxf(a, b);
      mx = fmaxf(mx, __shfl_xor(mx, 1, 32));
      mx = fmaxf(mx, __shfl_xor(mx, 2, 32));
      mx = fmaxf(mx, __shfl_xor(mx, 4, 32));
      mx = fmaxf(mx, __shfl_xor(mx, 8, 32));
      float mnew = fmaxf(mi[j], mx);
      float alpha = __expf(mi[j] - mnew);
      float p0 = __expf(a - mnew);
      float p1 = __expf(b - mnew);
      float rs = p0 + p1;
      rs += __shfl_xor(rs, 1, 32);
      rs += __shfl_xor(rs, 2, 32);
      rs += __shfl_xor(rs, 4, 32);
      rs += __shfl_xor(rs, 8, 32);
      li[j] = li[j] * alpha + rs;
      mi[j] = mnew;
#pragma unroll
      for (int c = 0; c < 8; ++c) oacc[c][j] *= alpha;
      int m = (j + 8 * gh) * 32;  // P[m][key], row stride 32 halves
      P[m + lr]      = (bf16)p0;
      P[m + 16 + lr] = (bf16)p1;
    }
    // ---- P (A-frag) @ V ------------------------------------------------
    v16bf pf = ldfrag(&P[lr * 32 + gh * 8]);  // same-wave DS ordering
#pragma unroll
    for (int c = 0; c < 8; ++c) {
      v16bf vf = ldfrag(&Vt[cur][(c * 16 + lr) * 32 + gh * 8]);
      oacc[c] = wmma_bf16(pf, vf, oacc[c]);
    }
  }

  // normalize and store: out[s][head*HD + d], f32
#pragma unroll
  for (int c = 0; c < 8; ++c)
#pragma unroll
    for (int j = 0; j < 8; ++j) {
      size_t row = qrow + j + 8 * gh;
      size_t col = head * HD_ + c * 16 + lr;
      o[row * H_ + col] = oacc[c][j] / li[j];
    }
}

// ---------------- Host-side orchestration -----------------------------------

extern "C" void kernel_launch(void* const* d_in, const int* in_sizes, int n_in,
                              void* d_out, int out_size, void* d_ws,
                              size_t ws_size, hipStream_t stream) {
  (void)in_sizes; (void)n_in; (void)out_size; (void)ws_size;
  const int*   ids    = (const int*)d_in[0];
  const float* emb    = (const float*)d_in[1];
  const float* Wq     = (const float*)d_in[2];
  const float* Wk     = (const float*)d_in[3];
  const float* Wv     = (const float*)d_in[4];
  const float* Wo     = (const float*)d_in[5];
  const float* Wg     = (const float*)d_in[6];
  const float* Wu     = (const float*)d_in[7];
  const float* Wd     = (const float*)d_in[8];
  const float* ln1    = (const float*)d_in[9];
  const float* ln2    = (const float*)d_in[10];
  const float* norm_w = (const float*)d_in[11];
  float* out = (float*)d_out;

  // workspace carve-up (all offsets 256B aligned)
  char* ws = (char*)d_ws;
  size_t off = 0;
  auto carve = [&](size_t bytes) { void* p = ws + off; off += (bytes + 255) & ~(size_t)255; return p; };
  float* x   = (float*)carve((size_t)S_ * H_ * 4);    // residual stream
  float* h   = (float*)carve((size_t)S_ * H_ * 4);    // rmsnorm out / attn out
  float* q   = (float*)carve((size_t)S_ * H_ * 4);    // q proj / h2
  float* k   = (float*)carve((size_t)S_ * KVW * 4);
  float* v   = (float*)carve((size_t)S_ * KVW * 4);
  float* g   = (float*)carve((size_t)S_ * I_ * 4);
  float* u   = (float*)carve((size_t)S_ * I_ * 4);
  bf16*  qbf = (bf16*)carve((size_t)S_ * H_ * 2);
  bf16*  kbf = (bf16*)carve((size_t)S_ * KVW * 2);
  bf16*  vbT = (bf16*)carve((size_t)S_ * KVW * 2);

  auto gemm = [&](const float* A, const float* B, const float* add, float* C,
                  int M, int N, int K) {
    dim3 grid(N / BN, M / BM);
    k_gemm<<<grid, 256, 0, stream>>>(A, B, add, C, M, N, K);
  };

  k_embed<<<S_, 256, 0, stream>>>(ids, emb, x);

  for (int l = 0; l < L_; ++l) {
    const float* wq = Wq + (size_t)l * H_ * H_;
    const float* wk = Wk + (size_t)l * H_ * KVW;
    const float* wv = Wv + (size_t)l * H_ * KVW;
    const float* wo = Wo + (size_t)l * H_ * H_;
    const float* wg = Wg + (size_t)l * H_ * I_;
    const float* wu = Wu + (size_t)l * H_ * I_;
    const float* wd = Wd + (size_t)l * I_ * H_;

    // attention block
    k_rmsnorm<<<S_, 256, 0, stream>>>(x, ln1 + (size_t)l * H_, h);
    gemm(h, wq, nullptr, q, S_, H_, H_);
    gemm(h, wk, nullptr, k, S_, KVW, H_);
    gemm(h, wv, nullptr, v, S_, KVW, H_);
    k_rope_q<<<(S_ * NH_ * 64) / 256, 256, 0, stream>>>(q, qbf);
    k_rope_k<<<(S_ * NKV_ * 64) / 256, 256, 0, stream>>>(k, kbf);
    k_vT<<<(S_ * KVW) / 256, 256, 0, stream>>>(v, vbT);
    k_attn<<<dim3(NH_, S_ / 16), 32, 0, stream>>>(qbf, kbf, vbT, h);
    gemm(h, wo, x, x, S_, H_, H_);  // x += attn_out @ Wo

    // MLP block (h2 reuses q buffer)
    k_rmsnorm<<<S_, 256, 0, stream>>>(x, ln2 + (size_t)l * H_, q);
    gemm(q, wg, nullptr, g, S_, I_, H_);
    gemm(q, wu, nullptr, u, S_, I_, H_);
    k_silumul<<<(size_t)S_ * I_ / 256, 256, 0, stream>>>(g, u);
    gemm(g, wd, x, x, S_, H_, I_);  // x += (silu(g)*u) @ Wd
  }

  k_rmsnorm<<<S_, 256, 0, stream>>>(x, norm_w, out);
}
